// MixHopConv_20469814133013
// MI455X (gfx1250) — matrix operand order repaired and checked by reading, not compiled, not run on hardware
//
#include <hip/hip_runtime.h>
#include <hip/hip_bf16.h>

// ---------------------------------------------------------------------------
// MixHop GIN on MI455X (gfx1250).
//  * Edge aggregation: L2-resident f32 atomic scatter (bound by L2 atomics,
//    everything fits in the 192MB L2; HBM 23.3TB/s never the limiter).
//  * All GEMMs: v_wmma_f32_16x16x32_bf16 (bf16 in, f32 accumulate).
//  * Weights pre-swizzled into WMMA B-fragment layout -> 1x 32B load / frag.
// ---------------------------------------------------------------------------

#define N_NODES 50000
#define HIDDEN  64
#define N_EDGES 800000
#define ND      (N_NODES * HIDDEN)

typedef __attribute__((ext_vector_type(16))) __bf16 v16bf;
typedef __attribute__((ext_vector_type(8)))  __bf16 v8bf;
typedef __attribute__((ext_vector_type(8)))  float  v8f;

__device__ __forceinline__ v8f wmma_bf16(v16bf a, v16bf b, v8f c) {
    // (neg_a, A, neg_b, B, c_mod, C, reuse_a, reuse_b)
    return __builtin_amdgcn_wmma_f32_16x16x32_bf16(false, a, false, b,
                                                   (short)0, c, false, false);
}

// Build a 16x32 bf16 A-fragment from f32 row-major data.
// ISA A layout (16-bit, 16x32): lane<16: row=lane, elems 0..7 = K k0..k0+7,
// elems 8..15 = K k0+16..k0+23; lane>=16: row=lane-16, same +8 K offset.
__device__ __forceinline__ v16bf load_a_f32(const float* __restrict__ src,
                                            int row, int ldm, int k0, int lane) {
    const float* p = src + (size_t)row * ldm + k0 + ((lane & 16) ? 8 : 0);
    v16bf a;
#pragma unroll
    for (int h = 0; h < 2; ++h) {
        const float4 u = *(const float4*)(p + h * 16);
        const float4 v = *(const float4*)(p + h * 16 + 4);
        a[h * 8 + 0] = (__bf16)u.x; a[h * 8 + 1] = (__bf16)u.y;
        a[h * 8 + 2] = (__bf16)u.z; a[h * 8 + 3] = (__bf16)u.w;
        a[h * 8 + 4] = (__bf16)v.x; a[h * 8 + 5] = (__bf16)v.y;
        a[h * 8 + 6] = (__bf16)v.z; a[h * 8 + 7] = (__bf16)v.w;
    }
    return a;
}

// Same A-fragment but from a bf16 16x64 tile in LDS (wave-private).
__device__ __forceinline__ v16bf load_a_lds(const __bf16* t, int k0, int lane) {
    const __bf16* p = t + (size_t)(lane & 15) * HIDDEN + k0 + ((lane & 16) ? 8 : 0);
    v8bf lo = *(const v8bf*)p;
    v8bf hi = *(const v8bf*)(p + 16);
    v16bf a;
#pragma unroll
    for (int j = 0; j < 8; ++j) { a[j] = lo[j]; a[8 + j] = hi[j]; }
    return a;
}

// Pre-swizzled B fragment: frag (kt,nt) is 32 lanes x 16 bf16 contiguous.
__device__ __forceinline__ v16bf load_b(const __bf16* __restrict__ wf,
                                        int kt, int nt, int lane) {
    return *(const v16bf*)(wf + ((size_t)((kt * 4 + nt) * 32 + lane) << 4));
}

// ---------------------------------------------------------------------------
// Swizzle a KxHIDDEN f32 weight matrix into WMMA B-fragment layout (bf16).
// B layout (16-bit, 32x16): lane<16: col N=lane, elems 0..7 -> K=k0+j, 8..15 -> K=k0+8+j;
// lane>=16: col N=lane-16, same +8 K offset.
__global__ void prep_frags_kernel(const float* __restrict__ W,
                                  __bf16* __restrict__ wf, int K) {
    const int t = blockIdx.x * blockDim.x + threadIdx.x;
    const int nfrag = (K / 32) * (HIDDEN / 16);
    if (t >= nfrag * 32) return;
    const int lane = t & 31;
    const int f    = t >> 5;
    const int kt   = f >> 2;
    const int nt   = f & 3;
    const int kb   = kt * 32 + ((lane & 16) ? 8 : 0);
    const int n    = nt * 16 + (lane & 15);
    __bf16* d = wf + ((size_t)(f * 32 + lane) << 4);
#pragma unroll
    for (int j = 0; j < 16; ++j) d[j] = (__bf16)W[(size_t)(kb + j + (j & 8)) * HIDDEN + n];
}

// z = h  (GIN eps=0: z starts as the node's own feature)
__global__ void copy_f4_kernel(const float4* __restrict__ src,
                               float4* __restrict__ dst, int n) {
    int i = blockIdx.x * blockDim.x + threadIdx.x;
    if (i < n) dst[i] = src[i];
}

// z[dst] += h[src] over all edges; 4 threads per edge, 16 floats each.
__global__ void scatter_kernel(const float* __restrict__ h,
                               const long long* __restrict__ src,
                               const long long* __restrict__ dst,
                               float* __restrict__ z, int nedges) {
    const int tid = blockIdx.x * blockDim.x + threadIdx.x;
    const int e = tid >> 2;
    if (e >= nedges) return;
    const int c = (tid & 3) * 16;
    const int s = (int)src[e];
    const int d = (int)dst[e];
    const float4* hp = (const float4*)(h + (size_t)s * HIDDEN + c);
    float* zp = z + (size_t)d * HIDDEN + c;
#pragma unroll
    for (int q = 0; q < 4; ++q) {
        float4 v = hp[q];
        atomicAdd(zp + 4 * q + 0, v.x);
        atomicAdd(zp + 4 * q + 1, v.y);
        atomicAdd(zp + 4 * q + 2, v.z);
        atomicAdd(zp + 4 * q + 3, v.w);
    }
}

// Fused GIN MLP: out = relu(z@W1 + b1)@W2 + b2.  One wave per 16-row tile.
__global__ void __launch_bounds__(128)
gin_mlp_kernel(const float* __restrict__ z,
               const __bf16* __restrict__ wf1, const float* __restrict__ b1,
               const __bf16* __restrict__ wf2, const float* __restrict__ b2,
               float* __restrict__ out, int nrows) {
    __shared__ __align__(16) __bf16 t_lds[4][16][HIDDEN];
    const int lane   = threadIdx.x & 31;
    const int wave   = threadIdx.x >> 5;
    const int base   = (blockIdx.x * 4 + wave) * 16;
    if (base >= nrows) return;                 // wave-uniform -> EXEC stays all-1s
    const int rowL   = base + (lane & 15);
    const int rc     = rowL < nrows ? rowL : nrows - 1;
    const int nc     = lane & 15;
    const int rshift = (lane & 16) ? 8 : 0;    // C/D frag: elem i -> row i(+8)

    // ---- layer 1 ----
    v16bf a0 = load_a_f32(z, rc, HIDDEN, 0,  lane);
    v16bf a1 = load_a_f32(z, rc, HIDDEN, 32, lane);
#pragma unroll
    for (int nt = 0; nt < 4; ++nt) {
        v8f acc = {};
        acc = wmma_bf16(a0, load_b(wf1, 0, nt, lane), acc);
        acc = wmma_bf16(a1, load_b(wf1, 1, nt, lane), acc);
        const float bias = b1[nt * 16 + nc];
#pragma unroll
        for (int i = 0; i < 8; ++i) {
            float v = acc[i] + bias;
            v = v > 0.f ? v : 0.f;
            t_lds[wave][i + rshift][nt * 16 + nc] = (__bf16)v;
        }
    }
    // LDS tile is wave-private and DS ops are in-order per wave: no barrier.

    // ---- layer 2 ----
    v16bf c0 = load_a_lds(&t_lds[wave][0][0], 0,  lane);
    v16bf c1 = load_a_lds(&t_lds[wave][0][0], 32, lane);
#pragma unroll
    for (int nt = 0; nt < 4; ++nt) {
        v8f acc = {};
        acc = wmma_bf16(c0, load_b(wf2, 0, nt, lane), acc);
        acc = wmma_bf16(c1, load_b(wf2, 1, nt, lane), acc);
        const float bias = b2[nt * 16 + nc];
#pragma unroll
        for (int i = 0; i < 8; ++i) {
            const int r = base + i + rshift;
            if (r < nrows)
                out[(size_t)r * HIDDEN + nt * 16 + nc] = acc[i] + bias;
        }
    }
}

// out = concat(h1,h2,h3) @ Wo + bo   (K=192 -> 6 K-tiles over 3 buffers)
__global__ void __launch_bounds__(128)
final_kernel(const float* __restrict__ h1, const float* __restrict__ h2,
             const float* __restrict__ h3, const __bf16* __restrict__ wfo,
             const float* __restrict__ bo, float* __restrict__ out, int nrows) {
    const int lane   = threadIdx.x & 31;
    const int wave   = threadIdx.x >> 5;
    const int base   = (blockIdx.x * 4 + wave) * 16;
    if (base >= nrows) return;
    const int rowL   = base + (lane & 15);
    const int rc     = rowL < nrows ? rowL : nrows - 1;
    const int nc     = lane & 15;
    const int rshift = (lane & 16) ? 8 : 0;
    const float* hops[3] = {h1, h2, h3};

    v8f acc[4] = {};
#pragma unroll
    for (int kt = 0; kt < 6; ++kt) {
        const float* src = hops[kt >> 1];
        v16bf a = load_a_f32(src, rc, HIDDEN, (kt & 1) * 32, lane);
#pragma unroll
        for (int nt = 0; nt < 4; ++nt)
            acc[nt] = wmma_bf16(a, load_b(wfo, kt, nt, lane), acc[nt]);
    }
#pragma unroll
    for (int nt = 0; nt < 4; ++nt) {
        const float bias = bo[nt * 16 + nc];
#pragma unroll
        for (int i = 0; i < 8; ++i) {
            const int r = base + i + rshift;
            if (r < nrows)
                out[(size_t)r * HIDDEN + nt * 16 + nc] = acc[nt][i] + bias;
        }
    }
}

// ---------------------------------------------------------------------------
static void run_conv(const float* hin, const __bf16* wf1, const float* b1,
                     const __bf16* wf2, const float* b2, float* hout,
                     const long long* src, const long long* dst, float* z,
                     hipStream_t stream) {
    copy_f4_kernel<<<(ND / 4 + 255) / 256, 256, 0, stream>>>(
        (const float4*)hin, (float4*)z, ND / 4);
    scatter_kernel<<<(N_EDGES * 4 + 255) / 256, 256, 0, stream>>>(
        hin, src, dst, z, N_EDGES);
    const int grid = (((N_NODES + 15) / 16) + 3) / 4;
    gin_mlp_kernel<<<grid, 128, 0, stream>>>(z, wf1, b1, wf2, b2, hout, N_NODES);
}

extern "C" void kernel_launch(void* const* d_in, const int* in_sizes, int n_in,
                              void* d_out, int out_size, void* d_ws, size_t ws_size,
                              hipStream_t stream) {
    (void)in_sizes; (void)n_in; (void)out_size; (void)ws_size;
    const float*     x   = (const float*)d_in[0];
    const long long* ei  = (const long long*)d_in[1];   // int64 per reference
    const long long* src = ei;
    const long long* dst = ei + N_EDGES;
    const float* W1[3] = {(const float*)d_in[2], (const float*)d_in[6],  (const float*)d_in[10]};
    const float* B1[3] = {(const float*)d_in[3], (const float*)d_in[7],  (const float*)d_in[11]};
    const float* W2[3] = {(const float*)d_in[4], (const float*)d_in[8],  (const float*)d_in[12]};
    const float* B2[3] = {(const float*)d_in[5], (const float*)d_in[9],  (const float*)d_in[13]};
    const float* Wo = (const float*)d_in[14];
    const float* bo = (const float*)d_in[15];
    float* out = (float*)d_out;

    // Workspace layout: 4 f32 node buffers + bf16 weight fragments.
    float* ws = (float*)d_ws;
    float* h1 = ws;
    float* h2 = h1 + ND;
    float* h3 = h2 + ND;
    float* z  = h3 + ND;
    __bf16* wf = (__bf16*)(z + ND);          // 32B-aligned (offset 4*ND*4 bytes)
    __bf16* wfW1[3]; __bf16* wfW2[3];
    for (int i = 0; i < 3; ++i) {
        wfW1[i] = wf + (size_t)i * 8192;     // 8 frags * 512 bf16 per 64x64 matrix
        wfW2[i] = wf + (size_t)i * 8192 + 4096;
    }
    __bf16* wfo = wf + 3 * 8192;             // 24 frags * 512 bf16 for 192x64 Wo

    // Swizzle all weights into WMMA B-fragment layout (tiny; once per launch).
    for (int i = 0; i < 3; ++i) {
        prep_frags_kernel<<<1, 256, 0, stream>>>(W1[i], wfW1[i], 64);
        prep_frags_kernel<<<1, 256, 0, stream>>>(W2[i], wfW2[i], 64);
    }
    prep_frags_kernel<<<3, 256, 0, stream>>>(Wo, wfo, 192);

    // hop 1: one conv with params 0
    run_conv(x,  wfW1[0], B1[0], wfW2[0], B2[0], h1, src, dst, z, stream);
    // hop 2: two convs with params 1
    run_conv(x,  wfW1[1], B1[1], wfW2[1], B2[1], h2, src, dst, z, stream);
    run_conv(h2, wfW1[1], B1[1], wfW2[1], B2[1], h2, src, dst, z, stream);
    // hop 3: three convs with params 2
    run_conv(x,  wfW1[2], B1[2], wfW2[2], B2[2], h3, src, dst, z, stream);
    run_conv(h3, wfW1[2], B1[2], wfW2[2], B2[2], h3, src, dst, z, stream);
    run_conv(h3, wfW1[2], B1[2], wfW2[2], B2[2], h3, src, dst, z, stream);

    // readout: concat(h1,h2,h3) @ Wo + bo
    const int grid = (((N_NODES + 15) / 16) + 3) / 4;
    final_kernel<<<grid, 128, 0, stream>>>(h1, h2, h3, wfo, bo, out, N_NODES);
}
